// DNCformerBlock_23759759081586
// MI455X (gfx1250) — matrix operand order repaired and checked by reading, not compiled, not run on hardware
//
#include <hip/hip_runtime.h>
#include <hip/hip_bf16.h>

#define B_ 8
#define T_ 512
#define DIN_ 512
#define DM_ 512
#define H_ 8
#define DH_ 64
#define R_ 4
#define W_ 64
#define N_ 128
#define IFOUT_ 471
#define MTOT_ (B_*T_)   // 4096 rows

typedef __attribute__((ext_vector_type(16))) __bf16 v16bf;
typedef __attribute__((ext_vector_type(8)))  float  v8f;

union bvec16 { v16bf v; __bf16 e[16]; unsigned u[8]; };
union fvec8  { v8f  v; float  e[8];  };

__device__ __forceinline__ float sigmoidf_(float x){ return 1.f/(1.f+__expf(-x)); }
__device__ __forceinline__ float softplusf_(float x){ return fmaxf(x,0.f) + log1pf(__expf(-fabsf(x))); }

// pack two fp32 -> bf16x2 in one dword (elem0 in low half, little-endian)
__device__ __forceinline__ unsigned pack_bf16(float a, float b){
    union { __bf16 h[2]; unsigned u; } c;
    c.h[0] = (__bf16)a; c.h[1] = (__bf16)b;
    return c.u;
}

// ---------------------------------------------------------------------------
// Generic bf16-WMMA GEMM: C[M,N] = epilogue(A[M,K] @ B[K,N] + bias[N])
// EPI: 0 = bias only, 1 = bias + exact GELU, 2 = bias + residual add
// Block tile 128x64, 8 waves (4x2), each wave 32x32 (2x2 WMMA tiles).
// All WMMA operands staged in LDS as packed bf16x2 pairs along K so each
// fragment VGPR is one aligned 32-bit LDS word (vectorizes to ds_load_b128).
// ---------------------------------------------------------------------------
template<int EPI>
__global__ void __launch_bounds__(256)
gemm_bias(const float* __restrict__ A, const float* __restrict__ Bw,
          const float* __restrict__ bias, const float* __restrict__ Resid,
          float* __restrict__ C, int M, int N, int K)
{
    // stride 20 dwords = 80B: rows stay 16B-aligned, fragment pattern is
    // bank-conflict-free (lr*20 mod 64 distinct for lr=0..15).
    __shared__ unsigned AsP[128][20];   // [row][kpair], 16 pairs = K32
    __shared__ unsigned BsP[64][20];    // [col][kpair]

    const int tid  = threadIdx.x;
    const int lane = tid & 31;
    const int wid  = tid >> 5;
    const int half = lane >> 4;
    const int lr   = lane & 15;
    const int wm   = wid & 3;     // which 32-row strip
    const int wn   = wid >> 2;    // which 32-col strip
    const int m0   = blockIdx.y * 128;
    const int n0   = blockIdx.x * 64;

    fvec8 acc[2][2];
    for (int i=0;i<2;++i) for (int j=0;j<2;++j) for (int e=0;e<8;++e) acc[i][j].e[e]=0.f;

    for (int k0=0; k0<K; k0+=32){
        // stage A tile: 128 rows x 16 kpairs, 8 dwords/thread, contiguous 64B gl reads
        {
            int r  = tid >> 1;
            int p0 = (tid & 1) * 8;
            int gr = m0 + r;
            if (gr < M){
                const float* src = A + (size_t)gr*K + k0 + p0*2;
                #pragma unroll
                for (int c=0;c<8;++c) AsP[r][p0+c] = pack_bf16(src[2*c], src[2*c+1]);
            } else {
                #pragma unroll
                for (int c=0;c<8;++c) AsP[r][p0+c] = 0u;
            }
        }
        // stage B tile: 64 cols x 16 kpairs, 4 dwords/thread (coalesced over cols)
        {
            int col = tid & 63;
            int p0  = (tid >> 6) * 4;
            int gc  = n0 + col;
            #pragma unroll
            for (int c=0;c<4;++c){
                int kp = p0 + c;
                unsigned val = 0u;
                if (gc < N)
                    val = pack_bf16(Bw[(size_t)(k0+2*kp)*N + gc],
                                    Bw[(size_t)(k0+2*kp+1)*N + gc]);
                BsP[col][kp] = val;
            }
        }
        // prefetch next K tile into caches (gfx1250 global_prefetch path)
        if (k0 + 32 < K){
            __builtin_prefetch(A + (size_t)(m0 + (tid>>1))*K + k0 + 32, 0, 0);
            __builtin_prefetch(Bw + (size_t)(k0 + 32 + (tid>>3))*N + n0, 0, 0);
        }
        __syncthreads();

        bvec16 af[2], bf2[2];
        #pragma unroll
        for (int i=0;i<2;++i){
            int row = wm*32 + i*16 + lr;
            #pragma unroll
            for (int v2=0; v2<4; ++v2){
                af[i].u[v2]   = AsP[row][half*4 + v2];       // K pairs 0..3 / 4..7
                af[i].u[v2+4] = AsP[row][half*4 + 8 + v2];   // K pairs 8..15
            }
        }
        #pragma unroll
        for (int j=0;j<2;++j){
            int col = wn*32 + j*16 + lr;
            #pragma unroll
            for (int v2=0; v2<8; ++v2)
                bf2[j].u[v2] = BsP[col][half*4 + (v2&3) + (v2>>2)*8];
        }
        #pragma unroll
        for (int i=0;i<2;++i)
            #pragma unroll
            for (int j=0;j<2;++j)
                acc[i][j].v = __builtin_amdgcn_wmma_f32_16x16x32_bf16(
                    false, af[i].v, false, bf2[j].v, (short)0, acc[i][j].v, false, false);
        __syncthreads();
    }

    // epilogue + store (C/D layout: row = v + 8*half, col = lr within 16x16)
    #pragma unroll
    for (int i=0;i<2;++i)
        #pragma unroll
        for (int j=0;j<2;++j)
            #pragma unroll
            for (int v2=0; v2<8; ++v2){
                int row = m0 + wm*32 + i*16 + half*8 + v2;
                int col = n0 + wn*32 + j*16 + lr;
                if (row < M && col < N){
                    float val = acc[i][j].e[v2] + bias[col];
                    if (EPI == 1){
                        val = 0.5f*val*(1.f + erff(val*0.70710678118654752f));
                    } else if (EPI == 2){
                        val += Resid[(size_t)row*N + col];
                    }
                    C[(size_t)row*N + col] = val;
                }
            }
}

// ---------------------------------------------------------------------------
// Row LayerNorm over DM=512 (one block per row, 2 elems/thread)
// ---------------------------------------------------------------------------
__global__ void __launch_bounds__(256)
ln_kernel(const float* __restrict__ X, const float* __restrict__ g,
          const float* __restrict__ bta, float* __restrict__ Y)
{
    const int row = blockIdx.x;
    const int tid = threadIdx.x;
    const float* xr = X + (size_t)row*DM_;
    float v0 = xr[tid], v1 = xr[tid+256];
    __shared__ float red[256];
    red[tid] = v0 + v1;
    __syncthreads();
    for (int o=128;o>0;o>>=1){ if (tid<o) red[tid]+=red[tid+o]; __syncthreads(); }
    float mean = red[0] * (1.f/512.f);
    __syncthreads();
    float d0 = v0-mean, d1 = v1-mean;
    red[tid] = d0*d0 + d1*d1;
    __syncthreads();
    for (int o=128;o>0;o>>=1){ if (tid<o) red[tid]+=red[tid+o]; __syncthreads(); }
    float inv = rsqrtf(red[0]*(1.f/512.f) + 1e-5f);
    Y[(size_t)row*DM_ + tid]     = d0*inv*g[tid]     + bta[tid];
    Y[(size_t)row*DM_ + tid+256] = d1*inv*g[tid+256] + bta[tid+256];
}

// ---------------------------------------------------------------------------
// Flash-style causal attention, WMMA for QK^T and PV, packed bf16x2 staging.
// grid = B*H*8 blocks; block handles one 64-row q tile of one (b,h).
// ---------------------------------------------------------------------------
__global__ void __launch_bounds__(256)
attn_kernel(const float* __restrict__ qkv, float* __restrict__ aout)
{
    const int qi = blockIdx.x & 7;
    const int bh = blockIdx.x >> 3;
    const int h  = bh & 7;
    const int b  = bh >> 3;
    const int tid  = threadIdx.x;
    const int lane = tid & 31, wid = tid >> 5;
    const int half = lane >> 4, lr = lane & 15;
    const int wm   = wid & 3,  wn = wid >> 2;

    // strides of 36 dwords = 144B (16B aligned rows)
    __shared__ unsigned QsP[64][36];   // [qrow][d-pair]   (A for QK^T)
    __shared__ unsigned KtP[64][36];   // [key][d-pair]    (B for QK^T)
    __shared__ unsigned VtP[64][36];   // [d][key-pair]    (B for PV)
    __shared__ unsigned PsP[64][36];   // [qrow][key-pair] (A for PV)
    __shared__ float    Ss[64][66];
    __shared__ float mrow[64], lrow[64], arow[64];

    const int q0 = qi*64;
    {   // load Q tile packed, fold in 1/sqrt(DH)=0.125 (exact power of two)
        int r = tid >> 2, p0 = (tid & 3)*8;
        const float* src = qkv + ((size_t)(b*T_ + q0 + r))*(3*DM_) + h*DH_ + p0*2;
        #pragma unroll
        for (int c=0;c<8;++c) QsP[r][p0+c] = pack_bf16(src[2*c]*0.125f, src[2*c+1]*0.125f);
    }
    if (tid < 64){ mrow[tid] = -3.0e38f; lrow[tid] = 0.f; }
    fvec8 oacc[2];
    for (int s=0;s<2;++s) for (int e=0;e<8;++e) oacc[s].e[e]=0.f;
    __syncthreads();

    for (int j=0; j<=qi; ++j){
        {   // K: pack d-pairs straight from row-major global (no transpose needed)
            int r = tid >> 2, p0 = (tid & 3)*8;
            const float* ksrc = qkv + ((size_t)(b*T_ + j*64 + r))*(3*DM_) + DM_ + h*DH_ + p0*2;
            #pragma unroll
            for (int c=0;c<8;++c) KtP[r][p0+c] = pack_bf16(ksrc[2*c], ksrc[2*c+1]);
        }
        {   // V: [d][key-pair], key pairs packed from two global rows
            int d  = tid & 63;
            int p0 = (tid >> 6) * 8;
            const float* vb = qkv + ((size_t)(b*T_ + j*64))*(3*DM_) + 2*DM_ + h*DH_ + d;
            #pragma unroll
            for (int c=0;c<8;++c){
                int kp = p0 + c;
                VtP[d][kp] = pack_bf16(vb[(size_t)(2*kp)*(3*DM_)],
                                       vb[(size_t)(2*kp+1)*(3*DM_)]);
            }
        }
        __syncthreads();

        // S = Q @ K^T  (64x64, contraction over DH=64 in two 32-chunks)
        fvec8 sacc[2];
        for (int s=0;s<2;++s) for (int e=0;e<8;++e) sacc[s].e[e]=0.f;
        #pragma unroll
        for (int kc=0;kc<2;++kc){
            bvec16 af;
            int row = wm*16 + lr;
            #pragma unroll
            for (int v2=0;v2<4;++v2){
                af.u[v2]   = QsP[row][half*4 + v2     + kc*16];
                af.u[v2+4] = QsP[row][half*4 + 8 + v2 + kc*16];
            }
            #pragma unroll
            for (int s=0;s<2;++s){
                bvec16 bf2;
                int col = wn*32 + s*16 + lr;
                #pragma unroll
                for (int v2=0;v2<8;++v2)
                    bf2.u[v2] = KtP[col][half*4 + (v2&3) + (v2>>2)*8 + kc*16];
                sacc[s].v = __builtin_amdgcn_wmma_f32_16x16x32_bf16(
                    false, af.v, false, bf2.v, (short)0, sacc[s].v, false, false);
            }
        }
        // write masked S to LDS
        #pragma unroll
        for (int s=0;s<2;++s)
            #pragma unroll
            for (int v2=0;v2<8;++v2){
                int rl = wm*16 + half*8 + v2;
                int cl = wn*32 + s*16 + lr;
                int qg = q0 + rl, kg = j*64 + cl;
                Ss[rl][cl] = (kg <= qg) ? sacc[s].e[v2] : -1e30f;
            }
        __syncthreads();

        // online softmax: one thread per q-row, writes P already pair-packed
        if (tid < 64){
            float mx = -3.0e38f;
            for (int c=0;c<64;++c) mx = fmaxf(mx, Ss[tid][c]);
            float mold = mrow[tid];
            float mnew = fmaxf(mold, mx);
            float alpha = __expf(mold - mnew);
            float sm = 0.f;
            for (int c=0;c<32;++c){
                float p0 = __expf(Ss[tid][2*c]  -mnew);
                float p1 = __expf(Ss[tid][2*c+1]-mnew);
                PsP[tid][c] = pack_bf16(p0, p1);
                sm += p0 + p1;
            }
            lrow[tid] = lrow[tid]*alpha + sm;
            mrow[tid] = mnew;
            arow[tid] = alpha;
        }
        __syncthreads();

        // rescale accumulators, then O += P @ V
        #pragma unroll
        for (int s=0;s<2;++s)
            #pragma unroll
            for (int v2=0;v2<8;++v2){
                int rl = wm*16 + half*8 + v2;
                oacc[s].e[v2] *= arow[rl];
            }
        #pragma unroll
        for (int kc=0;kc<2;++kc){
            bvec16 af;
            int row = wm*16 + lr;
            #pragma unroll
            for (int v2=0;v2<4;++v2){
                af.u[v2]   = PsP[row][half*4 + v2     + kc*16];
                af.u[v2+4] = PsP[row][half*4 + 8 + v2 + kc*16];
            }
            #pragma unroll
            for (int s=0;s<2;++s){
                bvec16 bf2;
                int col = wn*32 + s*16 + lr;
                #pragma unroll
                for (int v2=0;v2<8;++v2)
                    bf2.u[v2] = VtP[col][half*4 + (v2&3) + (v2>>2)*8 + kc*16];
                oacc[s].v = __builtin_amdgcn_wmma_f32_16x16x32_bf16(
                    false, af.v, false, bf2.v, (short)0, oacc[s].v, false, false);
            }
        }
        __syncthreads();
    }

    // normalize and store into (B,T,DM) activation layout
    #pragma unroll
    for (int s=0;s<2;++s)
        #pragma unroll
        for (int v2=0;v2<8;++v2){
            int rl = wm*16 + half*8 + v2;
            int cl = wn*32 + s*16 + lr;
            aout[((size_t)(b*T_ + q0 + rl))*DM_ + h*DH_ + cl] = oacc[s].e[v2] / lrow[rl];
        }
}

// ---------------------------------------------------------------------------
// DNC memory scan: one block per batch, sequential over T steps.
// M and small state in LDS; L (128x128) per batch in global (lives in L2).
// ---------------------------------------------------------------------------
__global__ void __launch_bounds__(256)
dnc_scan(const float* __restrict__ vif, float* __restrict__ reads,
         float* __restrict__ Lg)
{
    const int b   = blockIdx.x;
    const int tid = threadIdx.x;
    float* L = Lg + (size_t)b * (N_*N_);

    __shared__ float Ms[N_][W_];
    __shared__ float u_s[N_], p_s[N_], ww_s[N_], wwn[N_], pprev[N_];
    __shared__ float rw_s[R_][N_], rwn[R_][N_];
    __shared__ float cw_s[N_], mnorm[N_];
    __shared__ float cr_s[R_][N_], fwd[R_][N_], bwd[R_][N_];
    __shared__ float skey[N_]; __shared__ int sidx[N_];
    __shared__ float asort[N_], alloc_s[N_];
    __shared__ float kr_s[R_][W_], kw_s[W_], er_s[W_], wv_s[W_];
    __shared__ float br_s[R_], fg_s[R_], rm_s[R_][3], krn[R_];
    __shared__ float scal[8]; // 0:bw 1:ag 2:wg 3:|kw| 4:sum(ww)

    for (int i=tid;i<N_*W_;i+=256) Ms[i/W_][i%W_] = 0.f;
    for (int i=tid;i<N_;i+=256){ u_s[i]=0.f; p_s[i]=0.f; ww_s[i]=0.f; }
    for (int i=tid;i<R_*N_;i+=256) rw_s[i/N_][i%N_] = ((i%N_)==0) ? 1.f : 0.f;
    for (int i=tid;i<N_*N_;i+=256) L[i] = 0.f;
    __syncthreads();

    for (int t=0;t<T_;++t){
        const float* v = vif + ((size_t)(b*T_+t))*IFOUT_;
        // ---- unpack interface vector ----
        kr_s[tid/W_][tid%W_] = v[tid];                                  // all 256
        if (tid < W_){
            kw_s[tid] = v[260+tid];
            er_s[tid] = sigmoidf_(v[325+tid]);
            wv_s[tid] = v[389+tid];
        }
        if (tid < R_){
            br_s[tid] = softplusf_(v[256+tid]);
            fg_s[tid] = sigmoidf_(v[453+tid]);
        }
        if (tid < R_*3) rm_s[tid/3][tid%3] = v[459+tid];
        if (tid == 0){
            scal[0] = softplusf_(v[324]);   // beta_write
            scal[1] = sigmoidf_(v[457]);    // alloc gate
            scal[2] = sigmoidf_(v[458]);    // write gate
        }
        __syncthreads();

        // ---- usage update ----
        if (tid < N_){
            float uu = u_s[tid];
            float wo = ww_s[tid];
            uu = uu + (1.f-uu)*fminf(fmaxf(1.f-wo,0.f),1.f);
            float psi = 1.f;
            for (int r=0;r<R_;++r) psi *= (1.f - fg_s[r]*rw_s[r][tid]);
            u_s[tid] = fminf(fmaxf(uu*psi,0.f),1.f);
        }
        // ---- content write weights (cosine) ----
        if (tid < N_){
            float s=0.f;
            for (int w=0;w<W_;++w){ float m = Ms[tid][w]; s += m*m; }
            mnorm[tid] = fmaxf(sqrtf(s),1e-12f);
        }
        if (tid == 0){
            float s=0.f;
            for (int w=0;w<W_;++w) s += kw_s[w]*kw_s[w];
            scal[3] = fmaxf(sqrtf(s),1e-12f);
        }
        __syncthreads();
        if (tid < N_){
            float d=0.f;
            for (int w=0;w<W_;++w) d += Ms[tid][w]*kw_s[w];
            cw_s[tid] = d/(mnorm[tid]*scal[3]) * scal[0];
        }
        __syncthreads();
        if (tid == 0){  // softmax over N
            float mx=-1e30f; for (int n=0;n<N_;++n) mx=fmaxf(mx,cw_s[n]);
            float sm=0.f;
            for (int n=0;n<N_;++n){ float e=__expf(cw_s[n]-mx); cw_s[n]=e; sm+=e; }
            float inv=1.f/sm; for (int n=0;n<N_;++n) cw_s[n]*=inv;
        }
        // ---- allocation: stable ascending sort of usage ----
        if (tid < N_){ skey[tid] = 1e-6f + (1.f-1e-6f)*u_s[tid]; sidx[tid]=tid; }
        __syncthreads();
        for (int k=2;k<=N_;k<<=1){
            for (int j=k>>1;j>0;j>>=1){
                if (tid < N_){
                    int l = tid ^ j;
                    if (l > tid){
                        bool asc = ((tid & k)==0);
                        float a=skey[tid], bk=skey[l];
                        int ia=sidx[tid], ib=sidx[l];
                        bool greater = (a>bk) || (a==bk && ia>ib);
                        if (greater == asc){
                            skey[tid]=bk; skey[l]=a;
                            sidx[tid]=ib; sidx[l]=ia;
                        }
                    }
                }
                __syncthreads();
            }
        }
        if (tid == 0){
            float prod=1.f;
            for (int i=0;i<N_;++i){ asort[i]=(1.f-skey[i])*prod; prod*=skey[i]; }
        }
        __syncthreads();
        if (tid < N_) alloc_s[sidx[tid]] = asort[tid];
        __syncthreads();
        // ---- write weights ----
        if (tid < N_){
            float ag=scal[1], wg=scal[2];
            wwn[tid] = wg*(ag*alloc_s[tid] + (1.f-ag)*cw_s[tid]);
        }
        __syncthreads();
        if (tid == 0){ float s=0.f; for (int n=0;n<N_;++n) s+=wwn[n]; scal[4]=s; }
        if (tid < N_) pprev[tid] = p_s[tid];
        __syncthreads();
        // ---- memory write ----
        for (int i=tid;i<N_*W_;i+=256){
            int n=i/W_, w=i%W_;
            Ms[n][w] = Ms[n][w]*(1.f - wwn[n]*er_s[w]) + wwn[n]*wv_s[w];
        }
        if (tid < N_) p_s[tid] = (1.f - scal[4])*pprev[tid] + wwn[tid];
        __syncthreads();
        // ---- temporal link update (global, L2 resident) ----
        for (int i=tid;i<N_*N_;i+=256){
            int ii=i/N_, jj=i%N_;
            float Lv = L[i];
            Lv = (1.f - wwn[ii] - wwn[jj])*Lv + pprev[ii]*wwn[jj];
            if (ii==jj) Lv = 0.f;
            L[i] = Lv;
        }
        // ---- content read weights (post-write M) ----
        if (tid < N_){
            float s=0.f;
            for (int w=0;w<W_;++w){ float m=Ms[tid][w]; s += m*m; }
            mnorm[tid] = fmaxf(sqrtf(s),1e-12f);
        }
        if (tid < R_){
            float s=0.f;
            for (int w=0;w<W_;++w) s += kr_s[tid][w]*kr_s[tid][w];
            krn[tid] = fmaxf(sqrtf(s),1e-12f);
        }
        __syncthreads();
        for (int o=tid;o<R_*N_;o+=256){
            int r=o/N_, n=o%N_;
            float d=0.f;
            for (int w=0;w<W_;++w) d += Ms[n][w]*kr_s[r][w];
            cr_s[r][n] = d/(mnorm[n]*krn[r]) * br_s[r];
        }
        __syncthreads();
        if (tid < R_){
            float mx=-1e30f; for (int n=0;n<N_;++n) mx=fmaxf(mx,cr_s[tid][n]);
            float sm=0.f;
            for (int n=0;n<N_;++n){ float e=__expf(cr_s[tid][n]-mx); cr_s[tid][n]=e; sm+=e; }
            float inv=1.f/sm; for (int n=0;n<N_;++n) cr_s[tid][n]*=inv;
        }
        __syncthreads();
        // ---- forward / backward weights (use OLD rw) ----
        for (int o=tid;o<R_*N_;o+=256){
            int r=o/N_, m=o%N_;
            float f=0.f, bb=0.f;
            for (int n=0;n<N_;++n){
                float rv = rw_s[r][n];
                f  += rv * L[n*N_+m];
                bb += rv * L[m*N_+n];
            }
            fwd[r][m]=f; bwd[r][m]=bb;
        }
        __syncthreads();
        // ---- read-mode blend -> new rw ----
        for (int o=tid;o<R_*N_;o+=256){
            int r=o/N_, m=o%N_;
            float a0=rm_s[r][0], a1=rm_s[r][1], a2=rm_s[r][2];
            float mx=fmaxf(a0,fmaxf(a1,a2));
            float e0=__expf(a0-mx), e1=__expf(a1-mx), e2=__expf(a2-mx);
            float inv=1.f/(e0+e1+e2);
            rwn[r][m] = (e0*bwd[r][m] + e1*cr_s[r][m] + e2*fwd[r][m])*inv;
        }
        __syncthreads();
        for (int o=tid;o<R_*N_;o+=256) rw_s[o/N_][o%N_] = rwn[o/N_][o%N_];
        if (tid < N_) ww_s[tid] = wwn[tid];
        __syncthreads();
        // ---- read vectors ----
        {
            int r = tid / W_, w = tid % W_;
            float s=0.f;
            for (int n=0;n<N_;++n) s += rw_s[r][n]*Ms[n][w];
            reads[((size_t)(b*T_+t))*(R_*W_) + tid] = s;
        }
        __syncthreads();
    }
}

// concat(h3[4096x512], reads[4096x256]) -> cat[4096x768]
__global__ void __launch_bounds__(256)
concat_kernel(const float* __restrict__ h3, const float* __restrict__ reads,
              float* __restrict__ cat)
{
    int idx = blockIdx.x*256 + threadIdx.x;
    if (idx >= MTOT_*768) return;
    int row = idx / 768, col = idx % 768;
    cat[idx] = (col < DM_) ? h3[(size_t)row*DM_ + col]
                           : reads[(size_t)row*(R_*W_) + (col - DM_)];
}

// ---------------------------------------------------------------------------
extern "C" void kernel_launch(void* const* d_in, const int* in_sizes, int n_in,
                              void* d_out, int out_size, void* d_ws, size_t ws_size,
                              hipStream_t stream)
{
    (void)in_sizes; (void)n_in; (void)out_size; (void)ws_size;
    const float* x     = (const float*)d_in[0];
    const float* Wp_in = (const float*)d_in[1];
    const float* bp_in = (const float*)d_in[2];
    const float* ln1_g = (const float*)d_in[3];
    const float* ln1_b = (const float*)d_in[4];
    const float* Wqkv  = (const float*)d_in[5];
    const float* bqkv  = (const float*)d_in[6];
    const float* Wo    = (const float*)d_in[7];
    const float* bo    = (const float*)d_in[8];
    const float* ln2_g = (const float*)d_in[9];
    const float* ln2_b = (const float*)d_in[10];
    const float* Wff1  = (const float*)d_in[11];
    const float* bff1  = (const float*)d_in[12];
    const float* Wff2  = (const float*)d_in[13];
    const float* bff2  = (const float*)d_in[14];
    const float* Wif   = (const float*)d_in[15];
    const float* bif   = (const float*)d_in[16];
    const float* Wout  = (const float*)d_in[17];
    const float* bout  = (const float*)d_in[18];

    float* ws = (float*)d_ws;
    const size_t o_h0    = 0;
    const size_t o_qkv   = o_h0    + (size_t)MTOT_*DM_;
    const size_t o_a     = o_qkv   + (size_t)MTOT_*3*DM_;
    const size_t o_h1    = o_a     + (size_t)MTOT_*DM_;
    const size_t o_h2    = o_h1    + (size_t)MTOT_*DM_;
    const size_t o_ff1   = o_h2    + (size_t)MTOT_*DM_;
    const size_t o_h3    = o_ff1   + (size_t)MTOT_*4*DM_;
    const size_t o_vif   = o_h3    + (size_t)MTOT_*DM_;
    const size_t o_reads = o_vif   + (size_t)MTOT_*IFOUT_;
    const size_t o_cat   = o_reads + (size_t)MTOT_*R_*W_;
    const size_t o_L     = o_cat   + (size_t)MTOT_*768;

    float* h0    = ws + o_h0;
    float* qkvb  = ws + o_qkv;
    float* ab    = ws + o_a;
    float* h1    = ws + o_h1;
    float* h2    = ws + o_h2;
    float* ff1   = ws + o_ff1;
    float* h3    = ws + o_h3;
    float* vifb  = ws + o_vif;
    float* readb = ws + o_reads;
    float* catb  = ws + o_cat;
    float* Lb    = ws + o_L;

    dim3 blk(256);

    // 1) h0 = x @ Wp_in[:512,:] + bp_in   (zero-padded ctrl rows contribute 0)
    gemm_bias<0><<<dim3(DM_/64, MTOT_/128), blk, 0, stream>>>(
        x, Wp_in, bp_in, nullptr, h0, MTOT_, DM_, DIN_);
    // 2) h0 = LN1(h0), in place
    ln_kernel<<<MTOT_, blk, 0, stream>>>(h0, ln1_g, ln1_b, h0);
    // 3) qkv = h0 @ Wqkv + bqkv
    gemm_bias<0><<<dim3((3*DM_)/64, MTOT_/128), blk, 0, stream>>>(
        h0, Wqkv, bqkv, nullptr, qkvb, MTOT_, 3*DM_, DM_);
    // 4) causal flash attention
    attn_kernel<<<B_*H_*(T_/64), blk, 0, stream>>>(qkvb, ab);
    // 5) h1 = h0 + a @ Wo + bo
    gemm_bias<2><<<dim3(DM_/64, MTOT_/128), blk, 0, stream>>>(
        ab, Wo, bo, h0, h1, MTOT_, DM_, DM_);
    // 6) h2 = LN2(h1)
    ln_kernel<<<MTOT_, blk, 0, stream>>>(h1, ln2_g, ln2_b, h2);
    // 7) ff1 = gelu(h2 @ Wff1 + bff1)
    gemm_bias<1><<<dim3((4*DM_)/64, MTOT_/128), blk, 0, stream>>>(
        h2, Wff1, bff1, nullptr, ff1, MTOT_, 4*DM_, DM_);
    // 8) h3 = h2 + ff1 @ Wff2 + bff2
    gemm_bias<2><<<dim3(DM_/64, MTOT_/128), blk, 0, stream>>>(
        ff1, Wff2, bff2, h2, h3, MTOT_, DM_, 4*DM_);
    // 9) vif = h3 @ Wif + bif  (N=471, bounds-checked)
    gemm_bias<0><<<dim3((IFOUT_+63)/64, MTOT_/128), blk, 0, stream>>>(
        h3, Wif, bif, nullptr, vifb, MTOT_, IFOUT_, DM_);
    // 10) DNC memory scan
    dnc_scan<<<B_, blk, 0, stream>>>(vifb, readb, Lb);
    // 11) concat(h3, reads)
    concat_kernel<<<(MTOT_*768 + 255)/256, blk, 0, stream>>>(h3, readb, catb);
    // 12) out = cat @ Wout + bout
    gemm_bias<0><<<dim3(DM_/64, MTOT_/128), blk, 0, stream>>>(
        catb, Wout, bout, nullptr, (float*)d_out, MTOT_, DM_, 768);
}
